// CRF_25185688224710
// MI455X (gfx1250) — matrix operand order repaired and checked by reading, not compiled, not run on hardware
//
#include <hip/hip_runtime.h>
#include <hip/hip_bf16.h>
#include <hip/hip_fp16.h>

typedef _Float16 v16h __attribute__((ext_vector_type(16)));
typedef float    v8f  __attribute__((ext_vector_type(8)));

#define L_BOS 62
#define L_EOS 63
#define LDIM  64

// ---------------------------------------------------------------------------
// Kernel 1: gold-path scores. One wave32 per batch.
// ---------------------------------------------------------------------------
__global__ __launch_bounds__(128) void crf_scores_kernel(
    const float* __restrict__ em, const int* __restrict__ tags,
    const float* __restrict__ mask, const float* __restrict__ trans,
    float* __restrict__ scores, int B, int S)
{
  int b    = blockIdx.x * 4 + (threadIdx.x >> 5);
  int lane = threadIdx.x & 31;
  if (b >= B) return;
  const size_t base = (size_t)b * S;

  float acc = 0.f, msum = 0.f;
  for (int t = lane; t < S; t += 32) {
    float mk = mask[base + t];
    msum += mk;
    if (t >= 1) {
      int ct = tags[base + t];
      int pt = tags[base + t - 1];
      acc += mk * (em[(base + t) * LDIM + ct] + trans[pt * LDIM + ct]);
    }
  }
  for (int off = 16; off > 0; off >>= 1) {
    acc  += __shfl_down(acc,  off, 32);
    msum += __shfl_down(msum, off, 32);
  }
  if (lane == 0) {
    int t0   = tags[base];
    int last = (int)(msum + 0.5f) - 1;
    int lt   = tags[base + last];
    scores[b] = trans[L_BOS * LDIM + t0] + em[base * LDIM + t0] + acc +
                trans[lt * LDIM + L_EOS];
  }
}

// ---------------------------------------------------------------------------
// Kernel 2: forward scan (partition function) via f16 WMMA in prob domain.
// One block = 128 threads = 4 waves = one 16-batch tile; wave w owns j-tile w.
// ---------------------------------------------------------------------------
__global__ __launch_bounds__(128) void crf_forward_kernel(
    const float* __restrict__ em, const float* __restrict__ mask,
    const float* __restrict__ trans, float* __restrict__ part, int B, int S)
{
  __shared__ v16h  pAv[2 * 32];    // p in WMMA A-fragment layout (f16), 2KB
  __shared__ float qC[16 * 64];    // raw matmul output, batch-major, 4KB
  __shared__ float cAcc[16];       // per-batch accumulated log-scale
  _Float16* pA = (_Float16*)pAv;

  const int tid  = threadIdx.x;
  const int wv   = tid >> 5;       // wave id == j-tile
  const int lane = tid & 31;
  const int tile = blockIdx.x;

  // ---- register-resident B operands: expT[k, wv*16+n] in B-fragment layout
  const int n = lane & 15, kh = lane >> 4;
  v16h Bf0, Bf1;
#pragma unroll
  for (int h = 0; h < 16; ++h) {
    int j = wv * 16 + n;
    Bf0[h] = (_Float16)__expf(trans[(      kh * 16 + h) * LDIM + j]);
    Bf1[h] = (_Float16)__expf(trans[(32 + kh * 16 + h) * LDIM + j]);
  }

  // ---- phase-B mapping: thread -> (batch m, 8-wide j chunk)
  const int m  = tid >> 3;
  const int jc = tid & 7;
  const int jb = jc * 8;
  const int b  = tile * 16 + m;
  // A-fragment slot holding p[m, jb..jb+7] (contiguous 16 bytes in LDS)
  const int f        = jb >> 5;
  const int kk       = jb & 31;
  const int slotLane = m + 16 * ((kk >> 4) & 1);
  const int hb       = (kk & 8) ? 8 : 0;
  _Float16* slot = pA + f * 512 + slotLane * 16 + hb;

  const float* emB = em + ((size_t)b * S) * LDIM + jb;
  const float* mkB = mask + (size_t)b * S;

  for (int t = 0; t < S; ++t) {
    if (t > 0) {
      // ---------------- phase A: C = p @ expT (2 chained WMMAs) ----------
      v16h A0 = pAv[lane];
      v16h A1 = pAv[32 + lane];
      v8f c = {};
      c = __builtin_amdgcn_wmma_f32_16x16x32_f16(false, A0, false, Bf0,
                                                 (short)0, c, false, false);
      c = __builtin_amdgcn_wmma_f32_16x16x32_f16(false, A1, false, Bf1,
                                                 (short)0, c, false, false);
      const int cm = 8 * (lane >> 4);
      const int cj = wv * 16 + (lane & 15);
#pragma unroll
      for (int r = 0; r < 8; ++r) qC[(cm + r) * 64 + cj] = c[r];
      __syncthreads();
    }

    // ------------------ phase B: scale, renormalize, repack -------------
    const float* ep = emB + (size_t)t * LDIM;
    float q[8];
    if (t == 0) {
#pragma unroll
      for (int i = 0; i < 8; ++i)
        q[i] = __expf(trans[L_BOS * LDIM + jb + i] + ep[i]);
    } else {
#pragma unroll
      for (int i = 0; i < 8; ++i)
        q[i] = qC[m * 64 + jb + i] * __expf(ep[i]);
    }
    float mx = q[0];
#pragma unroll
    for (int i = 1; i < 8; ++i) mx = fmaxf(mx, q[i]);
    for (int off = 1; off < 8; off <<= 1)
      mx = fmaxf(mx, __shfl_xor(mx, off, 32));

    bool active = (t == 0) || (mkB[t] > 0.f);
    if (active) {
      float inv = 1.0f / mx;
      union { _Float16 h[8]; uint4 u; } pk;
#pragma unroll
      for (int i = 0; i < 8; ++i) pk.h[i] = (_Float16)(q[i] * inv);
      *(uint4*)slot = pk.u;
      if (jc == 0) {
        if (t == 0) cAcc[m]  = __logf(mx);
        else        cAcc[m] += __logf(mx);
      }
    }
    if (t + 1 < S) __builtin_prefetch(ep + LDIM, 0, 0);
    __syncthreads();
  }

  // ---- finalize: partition = c + log(sum_j p[j] * exp(T[j, EOS]))
  union { _Float16 h[8]; uint4 u; } pk;
  pk.u = *(const uint4*)slot;
  float s = 0.f;
#pragma unroll
  for (int i = 0; i < 8; ++i)
    s += (float)pk.h[i] * __expf(trans[(jb + i) * LDIM + L_EOS]);
  for (int off = 1; off < 8; off <<= 1) s += __shfl_xor(s, off, 32);
  if (jc == 0) part[b] = cAcc[m] + __logf(s);
}

// ---------------------------------------------------------------------------
// Kernel 3: NLL = -(sum_b scores[b] - partition[b])
// ---------------------------------------------------------------------------
__global__ __launch_bounds__(512) void crf_reduce_kernel(
    const float* __restrict__ scores, const float* __restrict__ part,
    float* __restrict__ out, int B)
{
  __shared__ float sm[512];
  int tid = threadIdx.x;
  float v = 0.f;
  for (int i = tid; i < B; i += 512) v += scores[i] - part[i];
  sm[tid] = v;
  __syncthreads();
  for (int s = 256; s > 0; s >>= 1) {
    if (tid < s) sm[tid] += sm[tid + s];
    __syncthreads();
  }
  if (tid == 0) out[0] = -sm[0];
}

// ---------------------------------------------------------------------------
extern "C" void kernel_launch(void* const* d_in, const int* in_sizes, int n_in,
                              void* d_out, int out_size, void* d_ws, size_t ws_size,
                              hipStream_t stream) {
  const float* em    = (const float*)d_in[0];
  const int*   tags  = (const int*)d_in[1];
  const float* mask  = (const float*)d_in[2];
  const float* trans = (const float*)d_in[3];
  float*       out   = (float*)d_out;

  const int S  = 1024;               // sequence length (reference)
  const int B  = in_sizes[2] / S;    // mask is B*S
  float* scores = (float*)d_ws;      // [B]
  float* part   = scores + B;        // [B]

  crf_scores_kernel<<<(B + 3) / 4, 128, 0, stream>>>(em, tags, mask, trans,
                                                     scores, B, S);
  crf_forward_kernel<<<B / 16, 128, 0, stream>>>(em, mask, trans, part, B, S);
  crf_reduce_kernel<<<1, 512, 0, stream>>>(scores, part, out, B);
}
